// AsymAttention_47802986004593
// MI455X (gfx1250) — matrix operand order repaired and checked by reading, not compiled, not run on hardware
//
#include <hip/hip_runtime.h>
#include <hip/hip_bf16.h>

typedef __bf16 bf16;
typedef __bf16 v16bf __attribute__((ext_vector_type(16)));
typedef float  v8f   __attribute__((ext_vector_type(8)));
typedef unsigned int u32x4 __attribute__((ext_vector_type(4)));
typedef int          i32x8 __attribute__((ext_vector_type(8)));
typedef int          i32x4 __attribute__((ext_vector_type(4)));

#define B_  2
#define N_  256
#define M_  64
#define D_  768
#define H_  12
#define HD_ 64

#if defined(__has_builtin)
#if __has_builtin(__builtin_amdgcn_tensor_load_to_lds) && \
    __has_builtin(__builtin_amdgcn_s_wait_tensorcnt)
#define HAVE_TDM 1
#endif
#endif
#ifndef HAVE_TDM
#define HAVE_TDM 0
#endif

// ---------------- WMMA helpers (wave32, 16x16x32 bf16 -> f32) ----------------

__device__ __forceinline__ v8f wmma_bf16(v16bf a, v16bf b, v8f c) {
  return __builtin_amdgcn_wmma_f32_16x16x32_bf16(false, a, false, b, (short)0, c,
                                                 false, false);
}

// A fragment: 16(M) x 32(K) from row-major bf16, leading dim ld, k-offset kk.
// Lanes 0-15: row=lane, elems 0..15 = K 0..15.
// Lanes 16-31: row=lane-16, elems 0..15 = K 16..31.
__device__ __forceinline__ v16bf load_a_bf16(const bf16* src, int ld, int kk, int lane) {
  const int m  = lane & 15;
  const int kb = (lane < 16) ? 0 : 16;
  const bf16* p = src + (size_t)m * ld + kk;
  v16bf a;
#pragma unroll
  for (int e = 0; e < 8; ++e) a[e] = p[kb + e];
#pragma unroll
  for (int e = 0; e < 8; ++e) a[8 + e] = p[8 + kb + e];
  return a;
}

// Same A fragment but converting from fp32 source (global or LDS).
__device__ __forceinline__ v16bf load_a_f32(const float* src, int ld, int kk, int lane) {
  const int m  = lane & 15;
  const int kb = (lane < 16) ? 0 : 16;
  const float* p = src + (size_t)m * ld + kk;
  v16bf a;
#pragma unroll
  for (int e = 0; e < 8; ++e) a[e] = (bf16)p[kb + e];
#pragma unroll
  for (int e = 0; e < 8; ++e) a[8 + e] = (bf16)p[8 + kb + e];
  return a;
}

// B fragment: 32(K) x 16(N) from transposed storage BT[col][k], leading dim ld.
// Lane = column (lane&15); lanes 0-15 hold K 0..15, lanes 16-31 hold K 16..31.
__device__ __forceinline__ v16bf load_bT_bf16(const bf16* srcT, int ld, int kk, int lane) {
  const int n  = lane & 15;
  const int kb = (lane < 16) ? 0 : 16;
  const bf16* p = srcT + (size_t)n * ld + kk + kb;
  v16bf b;
#pragma unroll
  for (int e = 0; e < 16; ++e) b[e] = p[e];
  return b;
}

// B fragment from row-major (K x N) storage: per-lane strided gather.
__device__ __forceinline__ v16bf load_b_strided(const bf16* src, int ld, int lane) {
  const int n  = lane & 15;
  const int kb = (lane < 16) ? 0 : 16;
  v16bf b;
#pragma unroll
  for (int e = 0; e < 16; ++e) b[e] = src[(size_t)(kb + e) * ld + n];
  return b;
}

__device__ __forceinline__ int cfrag_col(int lane)  { return lane & 15; }
__device__ __forceinline__ int cfrag_row0(int lane) { return (lane < 16) ? 0 : 8; }

// ---------------- Tensor Data Mover: 2D tile -> LDS ----------------
// Builds a D# (groups 0/1 per CDNA5 ISA) for a 2D tile of
// tile_h rows x tile_w elements, row stride `stride` elements, element size
// 2^elem_log2 bytes, and issues TENSOR_LOAD_TO_LDS. Tracked by TENSORcnt.
// This toolchain exposes the 6-arg builtin:
//   (uint32x4 g0, int32x8 g1, int32x4, int32x4, int32x8, i32 cpol)

#if HAVE_TDM
__device__ __forceinline__ void tdm_load_2d(unsigned lds_byte_addr, const void* gptr,
                                            unsigned elem_log2, unsigned tile_w,
                                            unsigned tile_h, unsigned stride_elems) {
  const unsigned long long ga = (unsigned long long)(size_t)gptr;
  u32x4 g0;
  g0[0] = 1u;                                                  // count=1, user, no gather
  g0[1] = lds_byte_addr;                                       // D#.lds_addr
  g0[2] = (unsigned)ga;                                        // global_addr[31:0]
  g0[3] = (unsigned)((ga >> 32) & 0x01ffffffu) | (2u << 30);   // addr[56:32] | type=2
  const unsigned td = 0x40000000u;                             // huge dims: no OOB clip
  i32x8 g1;
  g1[0] = (int)(elem_log2 << 16);                              // wg_mask=0 | data_size
  g1[1] = (int)((td & 0xffffu) << 16);                         // tensor_dim0[15:0]
  g1[2] = (int)((td >> 16) | ((td & 0xffffu) << 16));          // td0[31:16] | td1[15:0]
  g1[3] = (int)((td >> 16) | (tile_w << 16));                  // td1[31:16] | tile_dim0
  g1[4] = (int)(tile_h & 0xffffu);                             // tile_dim1 (tile_dim2=0)
  g1[5] = (int)stride_elems;                                   // dim0_stride[31:0]
  g1[6] = 0;                                                   // dim0_stride[47:32] | d1s
  g1[7] = 0;
  const i32x4 z4 = {0, 0, 0, 0};
  const i32x8 z8 = {0, 0, 0, 0, 0, 0, 0, 0};
  __builtin_amdgcn_tensor_load_to_lds(g0, g1, z4, z4, z8, 0);
}
#endif

// ---------------- Kernel 0: weight transpose + fp32->bf16 ----------------

__global__ void pack_kernel(const float* __restrict__ Wq, const float* __restrict__ Wkv,
                            const float* __restrict__ Wp,
                            bf16* __restrict__ WqT, bf16* __restrict__ WkvT,
                            bf16* __restrict__ WpT) {
  const int step = gridDim.x * blockDim.x;
  const int t0 = blockIdx.x * blockDim.x + threadIdx.x;
  const int n1 = D_ * D_;
  const int n2 = 2 * D_ * D_;
  for (int i = t0; i < n1; i += step) {
    const int c = i / D_, k = i % D_;
    WqT[i] = (bf16)Wq[k * D_ + c];
    WpT[i] = (bf16)Wp[k * D_ + c];
  }
  for (int i = t0; i < n2; i += step) {
    const int c = i / D_, k = i % D_;
    WkvT[i] = (bf16)Wkv[k * 2 * D_ + c];
  }
}

// ---------------- Kernel 1: q/k_x/v_x projection (WMMA) ----------------

__global__ __launch_bounds__(256) void qkv_kernel(
    const float* __restrict__ x, const float* __restrict__ bq,
    const float* __restrict__ bkv,
    const bf16* __restrict__ WqT, const bf16* __restrict__ WkvT,
    bf16* __restrict__ qv, bf16* __restrict__ kxv, bf16* __restrict__ vxv) {
  const int tid = threadIdx.x, wave = tid >> 5, lane = tid & 31;
  const int r0  = blockIdx.x * 16;
  const int gc0 = blockIdx.y * 128 + wave * 16;   // [0, 2304)

  const bf16* BT;
  const float* bias;
  if (gc0 < D_) { BT = WqT + (size_t)gc0 * D_;         bias = bq  + gc0; }
  else          { BT = WkvT + (size_t)(gc0 - D_) * D_; bias = bkv + (gc0 - D_); }

  v8f c = {};
  for (int ks = 0; ks < 24; ++ks) {
    const int kk = ks * 32;
    v16bf a  = load_a_f32(x + (size_t)r0 * D_, D_, kk, lane);
    v16bf bb = load_bT_bf16(BT, D_, kk, lane);
    c = wmma_bf16(a, bb, c);
  }
  const int n = cfrag_col(lane), mb = cfrag_row0(lane);
  const float bv = bias[n];
  const int gc = gc0 + n;
#pragma unroll
  for (int j = 0; j < 8; ++j) {
    const int r = r0 + mb + j;
    const int bb_ = r >> 8, nn = r & 255;
    float v = c[j] + bv;
    if (gc < D_) {                         // q (pre-scaled by HD^-0.5)
      v *= 0.125f;
      const int hh = gc >> 6, hd = gc & 63;
      qv[((size_t)(bb_ * H_ + hh) * N_ + nn) * HD_ + hd] = (bf16)v;
    } else {
      const int c2 = gc - D_;
      if (c2 < D_) {                       // k_x
        const int hh = c2 >> 6, hd = c2 & 63;
        kxv[((size_t)(bb_ * H_ + hh) * N_ + nn) * HD_ + hd] = (bf16)v;
      } else {                             // v_x
        const int c3 = c2 - D_;
        const int hh = c3 >> 6, hd = c3 & 63;
        vxv[((size_t)(bb_ * H_ + hh) * N_ + nn) * HD_ + hd] = (bf16)v;
      }
    }
  }
}

// ---------------- Fused attention kernel ----------------
// One WG per (b, h, n-tile of 16). k_s/v_s head slices are computed on the fly
// so kv_s (201 MB fp32) never touches HBM. Inner GEMM uses TDM double
// buffering: wave 0 issues tensor_load_to_lds for the NEXT k-panel (sim fp32
// 256x32 tile + WkvT bf16 64x32 tile) while all 8 waves run WMMA on the
// CURRENT panel; one s_wait_tensorcnt + barrier per k-step.
// LDS (dynamic, 130 KB):
//   q 2K | scores 16x320 f32 20K | out_loc 4K
//   | Apanel[2] 256x32 f32 (2x32K) | Bpanel[2] 64x32 bf16 (2x4K) | ksq 32K

#define SM_Q   0
#define SM_SC  2048
#define SM_OL  22528
#define SM_AP  26624u
#define SM_BP  92160u
#define SM_KS  100352
#define SM_TOT 133120

__device__ void local_gemm_quarter(const float* __restrict__ simQ,  // 256 x 768 fp32
                                   const bf16* __restrict__ BT,     // 64 x 768 (T)
                                   const float* __restrict__ bias,  // 64
                                   float* Apan, bf16* Bpan, bf16* ksq,
                                   int tid, int wave, int lane) {
  v8f acc[8];
#pragma unroll
  for (int i = 0; i < 8; ++i) { v8f z = {}; acc[i] = z; }

#if HAVE_TDM
  if (wave == 0) {
    tdm_load_2d(SM_AP, simQ, 2u, 32u, 256u, (unsigned)D_);
    tdm_load_2d(SM_BP, BT,   1u, 32u, 64u,  (unsigned)D_);
    __builtin_amdgcn_s_wait_tensorcnt(0);
  }
  __syncthreads();
  for (int ks = 0; ks < 24; ++ks) {
    const int cur = ks & 1;
    if (wave == 0 && (ks + 1) < 24) {
      const unsigned nxt = (unsigned)(cur ^ 1);
      tdm_load_2d(SM_AP + nxt * 32768u, simQ + (ks + 1) * 32, 2u, 32u, 256u, (unsigned)D_);
      tdm_load_2d(SM_BP + nxt * 4096u,  BT   + (ks + 1) * 32, 1u, 32u, 64u,  (unsigned)D_);
    }
    const float* Ab = Apan + cur * 8192;   // 256*32 floats per buffer
    const bf16*  Bb = Bpan + cur * 2048;   // 64*32 bf16 per buffer
#pragma unroll
    for (int rr = 0; rr < 2; ++rr) {
      const int rt = wave * 2 + rr;
      const v16bf a = load_a_f32(Ab + rt * 16 * 32, 32, 0, lane);
#pragma unroll
      for (int ct = 0; ct < 4; ++ct) {
        const v16bf bb = load_bT_bf16(Bb + ct * 16 * 32, 32, 0, lane);
        acc[rr * 4 + ct] = wmma_bf16(a, bb, acc[rr * 4 + ct]);
      }
    }
    if (wave == 0) __builtin_amdgcn_s_wait_tensorcnt(0);
    __syncthreads();
  }
#else
  for (int ks = 0; ks < 24; ++ks) {
    const int kk = ks * 32;
    {  // stage A panel (fp32): thread t owns sim row t of this quarter
      const float4* s4 = (const float4*)(simQ + (size_t)tid * D_ + kk);
      float4* dst = (float4*)(Apan + tid * 32);
#pragma unroll
      for (int i = 0; i < 8; ++i) dst[i] = s4[i];
      __builtin_prefetch((const void*)(simQ + (size_t)tid * D_ + kk + 32), 0, 0);
    }
    {  // stage B panel
      const int cl = tid >> 2, ko = (tid & 3) * 8;
      const bf16* src = BT + (size_t)cl * D_ + kk + ko;
      bf16* dst = Bpan + cl * 32 + ko;
#pragma unroll
      for (int i = 0; i < 8; ++i) dst[i] = src[i];
    }
    __syncthreads();
#pragma unroll
    for (int rr = 0; rr < 2; ++rr) {
      const int rt = wave * 2 + rr;
      const v16bf a = load_a_f32(Apan + rt * 16 * 32, 32, 0, lane);
#pragma unroll
      for (int ct = 0; ct < 4; ++ct) {
        const v16bf bb = load_bT_bf16(Bpan + ct * 16 * 32, 32, 0, lane);
        acc[rr * 4 + ct] = wmma_bf16(a, bb, acc[rr * 4 + ct]);
      }
    }
    __syncthreads();
  }
#endif

  // bias add + bf16 store to ksq
  const int n = cfrag_col(lane), mb = cfrag_row0(lane);
#pragma unroll
  for (int rr = 0; rr < 2; ++rr)
#pragma unroll
    for (int ct = 0; ct < 4; ++ct) {
      const float bv = bias[ct * 16 + n];
      const v8f cc = acc[rr * 4 + ct];
#pragma unroll
      for (int j = 0; j < 8; ++j) {
        const int row = (wave * 2 + rr) * 16 + mb + j;
        ksq[row * 64 + ct * 16 + n] = (bf16)(cc[j] + bv);
      }
    }
}

__global__ __launch_bounds__(256) void attn_kernel(
    const float* __restrict__ sim, const bf16* __restrict__ WkvT,
    const float* __restrict__ bkv,
    const bf16* __restrict__ qv, const bf16* __restrict__ kxv,
    const bf16* __restrict__ vxv, bf16* __restrict__ attn_out) {
  extern __shared__ char smem[];
  bf16*  q_lds   = (bf16*)(smem + SM_Q);
  float* scores  = (float*)(smem + SM_SC);
  float* out_loc = (float*)(smem + SM_OL);
  float* Apan    = (float*)(smem + SM_AP);
  bf16*  Bpan    = (bf16*)(smem + SM_BP);
  bf16*  ksq     = (bf16*)(smem + SM_KS);

  const int tid = threadIdx.x, wave = tid >> 5, lane = tid & 31;
  const int nt = blockIdx.x, h = blockIdx.y, b = blockIdx.z;
  const int n0 = nt * 16;
  const int bh = b * H_ + h;

  // phase 1: load q tile, zero local-out accumulator
  {
    const bf16* qsrc = qv + ((size_t)bh * N_ + n0) * HD_;
#pragma unroll
    for (int i = 0; i < 4; ++i) q_lds[tid * 4 + i] = qsrc[tid * 4 + i];
#pragma unroll
    for (int i = 0; i < 4; ++i) out_loc[tid * 4 + i] = 0.0f;
  }
  __syncthreads();

  // phase 2: global scores sg = q_scaled @ k_x^T  (16 x 256)
  {
    const int n = cfrag_col(lane), mb = cfrag_row0(lane);
#pragma unroll
    for (int g = 0; g < 2; ++g) {
      const int ct = wave + g * 8;
      v8f c = {};
#pragma unroll
      for (int ks = 0; ks < 2; ++ks) {
        const v16bf a  = load_a_bf16(q_lds, HD_, ks * 32, lane);
        const v16bf bb = load_bT_bf16(kxv + ((size_t)bh * N_ + ct * 16) * HD_,
                                      HD_, ks * 32, lane);
        c = wmma_bf16(a, bb, c);
      }
#pragma unroll
      for (int j = 0; j < 8; ++j)
        scores[(mb + j) * 320 + ct * 16 + n] = c[j];
    }
  }
  __syncthreads();

  // phase 3: local keys -> sl scores (4 quarters of 256 sim rows)
  const float* simb = sim + (size_t)(b * N_ + n0) * M_ * D_;
  for (int Q = 0; Q < 4; ++Q) {
    local_gemm_quarter(simb + (size_t)Q * 256 * D_,
                       WkvT + (size_t)(h * HD_) * D_, bkv + h * HD_,
                       Apan, Bpan, ksq, tid, wave, lane);
    __syncthreads();
    {  // thread t: row t of quarter -> (n_local = Q*4 + t/64, m = t%64)
      const int nl = Q * 4 + (tid >> 6), m = tid & 63;
      float s = 0.f;
#pragma unroll 8
      for (int hd = 0; hd < 64; ++hd)
        s += (float)q_lds[nl * 64 + hd] * (float)ksq[tid * 64 + hd];
      scores[nl * 320 + 256 + m] = s;
    }
    __syncthreads();
  }

  // phase 4: softmax over 320 (wave w owns rows 2w, 2w+1)
#pragma unroll
  for (int rr = 0; rr < 2; ++rr) {
    float* sr = scores + (wave * 2 + rr) * 320;
    float mx = -3.0e38f;
    for (int i = lane; i < 320; i += 32) mx = fmaxf(mx, sr[i]);
#pragma unroll
    for (int o = 16; o > 0; o >>= 1) mx = fmaxf(mx, __shfl_xor(mx, o, 32));
    float sum = 0.f;
    for (int i = lane; i < 320; i += 32) { const float e = __expf(sr[i] - mx); sr[i] = e; sum += e; }
#pragma unroll
    for (int o = 16; o > 0; o >>= 1) sum += __shfl_xor(sum, o, 32);
    const float inv = 1.0f / sum;
    for (int i = lane; i < 320; i += 32) sr[i] *= inv;
  }
  __syncthreads();

  // phase 5: local values -> out_loc += sum_m al * v_s
  for (int Q = 0; Q < 4; ++Q) {
    local_gemm_quarter(simb + (size_t)Q * 256 * D_,
                       WkvT + (size_t)(D_ + h * HD_) * D_, bkv + D_ + h * HD_,
                       Apan, Bpan, ksq, tid, wave, lane);
    __syncthreads();
    {
      const int nofs = tid >> 6, hd = tid & 63;
      const int nl = Q * 4 + nofs;
      const float* al = scores + nl * 320 + 256;
      float s = 0.f;
#pragma unroll 8
      for (int m = 0; m < 64; ++m)
        s += al[m] * (float)ksq[(nofs * 64 + m) * 64 + hd];
      out_loc[nl * 64 + hd] += s;
    }
    __syncthreads();
  }

  // phase 6: out = ag @ v_x + out_loc (waves 0..3, one 16-col tile each)
  if (wave < 4) {
    const int ct = wave;
    v8f c = {};
    for (int ks = 0; ks < 8; ++ks) {
      const int kk = ks * 32;
      const v16bf a  = load_a_f32(scores, 320, kk, lane);   // ag columns 0..255
      const v16bf bb = load_b_strided(vxv + ((size_t)bh * N_ + kk) * HD_ + ct * 16,
                                      HD_, lane);
      c = wmma_bf16(a, bb, c);
    }
    const int n = cfrag_col(lane), mb = cfrag_row0(lane);
#pragma unroll
    for (int j = 0; j < 8; ++j) {
      const int m = mb + j;
      const float v = c[j] + out_loc[m * 64 + ct * 16 + n];
      attn_out[(size_t)(b * N_ + n0 + m) * D_ + h * HD_ + ct * 16 + n] = (bf16)v;
    }
  }
}

// ---------------- Kernel 3: output projection (WMMA) ----------------

__global__ __launch_bounds__(256) void proj_kernel(
    const bf16* __restrict__ ao, const bf16* __restrict__ WpT,
    const float* __restrict__ bp, float* __restrict__ out) {
  const int tid = threadIdx.x, wave = tid >> 5, lane = tid & 31;
  const int r0 = blockIdx.x * 16;
  const int c0 = blockIdx.y * 128 + wave * 16;
  v8f c = {};
  for (int ks = 0; ks < 24; ++ks) {
    const int kk = ks * 32;
    const v16bf a  = load_a_bf16(ao + (size_t)r0 * D_, D_, kk, lane);
    const v16bf bb = load_bT_bf16(WpT + (size_t)c0 * D_, D_, kk, lane);
    c = wmma_bf16(a, bb, c);
  }
  const int n = cfrag_col(lane), mb = cfrag_row0(lane);
  const float bv = bp[c0 + n];
#pragma unroll
  for (int j = 0; j < 8; ++j)
    out[(size_t)(r0 + mb + j) * D_ + c0 + n] = c[j] + bv;
}

// ---------------- host launcher ----------------

extern "C" void kernel_launch(void* const* d_in, const int* in_sizes, int n_in,
                              void* d_out, int out_size, void* d_ws, size_t ws_size,
                              hipStream_t stream) {
  const float* x    = (const float*)d_in[0];
  const float* sim  = (const float*)d_in[1];
  const float* Wq   = (const float*)d_in[2];
  const float* bq   = (const float*)d_in[3];
  const float* Wkv  = (const float*)d_in[4];
  const float* bkv  = (const float*)d_in[5];
  const float* Wp   = (const float*)d_in[6];
  const float* bp   = (const float*)d_in[7];
  float* out = (float*)d_out;

  char* w = (char*)d_ws;
  bf16* WqT  = (bf16*)w; w += (size_t)D_ * D_ * 2;
  bf16* WkvT = (bf16*)w; w += (size_t)2 * D_ * D_ * 2;
  bf16* WpT  = (bf16*)w; w += (size_t)D_ * D_ * 2;
  bf16* qv   = (bf16*)w; w += (size_t)B_ * N_ * D_ * 2;
  bf16* kxv  = (bf16*)w; w += (size_t)B_ * N_ * D_ * 2;
  bf16* vxv  = (bf16*)w; w += (size_t)B_ * N_ * D_ * 2;
  bf16* ao   = (bf16*)w; w += (size_t)B_ * N_ * D_ * 2;   // total ~7.5 MB

  // Opt in to >64KB dynamic LDS for the fused attention kernel (config call,
  // not a stream operation: deterministic and graph-capture safe).
  (void)hipFuncSetAttribute((const void*)attn_kernel,
                            hipFuncAttributeMaxDynamicSharedMemorySize, SM_TOT);

  pack_kernel<<<1024, 256, 0, stream>>>(Wq, Wkv, Wp, WqT, WkvT, WpT);
  qkv_kernel<<<dim3(32, 18), 256, 0, stream>>>(x, bq, bkv, WqT, WkvT, qv, kxv, vxv);
  attn_kernel<<<dim3(16, 12, 2), 256, SM_TOT, stream>>>(sim, WkvT, bkv, qv, kxv, vxv, ao);
  proj_kernel<<<dim3(32, 6), 256, 0, stream>>>(ao, WpT, bp, out);
}
